// SPNN_41721312313478
// MI455X (gfx1250) — matrix operand order repaired and checked by reading, not compiled, not run on hardware
//
#include <hip/hip_runtime.h>

// ---------------------------------------------------------------------------
// MI455X (gfx1250) SPNN pipeline, wave32 + v_wmma_f32_16x16x32_bf16.
// Bandwidth-bound problem (213 GFLOP vs ~2.5 GB min traffic @ 23.3 TB/s),
// so: bf16 activation storage (halves dominant traffic), f32 WMMA accum,
// BN stats fused into GEMM epilogues (LDS reduce + atomics), BN+ReLU fused
// into next GEMM's LDS staging. node_feature (51MB) is L2-resident (192MB).
// Workspace use: 4 * E*64 bf16 ping-pong buffers (~1.02 GB) + tiny stats.
// ---------------------------------------------------------------------------

typedef __attribute__((ext_vector_type(16))) __bf16 v16bf;
typedef __attribute__((ext_vector_type(8)))  float  v8f;

#define FIN   224   // 3*H + G
#define TPAD  232   // layer-0 LDS row stride (bf16 elems) = 464B, 16B multiple
#define XPAD  72    // depth-layer LDS row stride (bf16 elems) = 144B

__device__ __forceinline__ unsigned short f2bfu(float f) {
  unsigned u = __builtin_bit_cast(unsigned, f);
  u += 0x7fffu + ((u >> 16) & 1u);          // round-to-nearest-even
  return (unsigned short)(u >> 16);
}
__device__ __forceinline__ float bf2f(unsigned short u) {
  unsigned x = ((unsigned)u) << 16;
  return __builtin_bit_cast(float, x);
}
struct Frag32B { uint4 lo, hi; };
__device__ __forceinline__ v16bf make_frag(uint4 lo, uint4 hi) {
  Frag32B t{lo, hi};
  return __builtin_bit_cast(v16bf, t);
}

// --------------------------- weight conversion -----------------------------
__global__ void spnn_convert_w(const float* __restrict__ W0, const float* __restrict__ Wd,
                               unsigned short* __restrict__ W0b, unsigned short* __restrict__ Wdb) {
  const int i = blockIdx.x * blockDim.x + threadIdx.x;
  if (i < 2 * 64 * FIN) W0b[i] = f2bfu(W0[i]);
  const int j = i - 2 * 64 * FIN;
  if (j >= 0 && j < 2 * 3 * 64 * 64) Wdb[j] = f2bfu(Wd[j]);
}

// --------------------------- mask population count -------------------------
__global__ void spnn_count(const int* __restrict__ sf, int E, unsigned* __restrict__ cnt) {
  __shared__ unsigned s;
  if (threadIdx.x == 0) s = 0;
  __syncthreads();
  unsigned local = 0;
  for (long long i = (long long)blockIdx.x * blockDim.x + threadIdx.x; i < E;
       i += (long long)gridDim.x * blockDim.x)
    local += (sf[i] != 0) ? 1u : 0u;
  atomicAdd(&s, local);
  __syncthreads();
  if (threadIdx.x == 0) atomicAdd(cnt, s);
}

// ------------------- layer 0: gather + GEMM (both branches) ----------------
// Block: 256 thr = 8 waves, 64-edge tile. Wave w -> rows [w*16, w*16+16).
__global__ __launch_bounds__(256) void spnn_gemm0(
    const float* __restrict__ nf, const float* __restrict__ geo,
    const int* __restrict__ eidx, const int* __restrict__ sf,
    const unsigned short* __restrict__ W0b, const float* __restrict__ b0,
    unsigned short* __restrict__ X0, unsigned short* __restrict__ X1,
    float* __restrict__ stats0, int E) {
  __shared__ alignas(16) unsigned short sA[64 * TPAD];
  __shared__ unsigned char sMask[64];     // 0=invalid 1=same_face 2=!same_face
  __shared__ float sStats[256];           // [br][sum|sumsq][64]

  const int tid = threadIdx.x;
  const long long tileBase = (long long)blockIdx.x * 64;
  sStats[tid] = 0.f;

  // stage 64 rows x 224 (f32 gathers -> bf16 LDS), 4 elems per thread-step
  for (int idx = tid; idx < 64 * 56; idx += 256) {
    const int r = idx / 56;
    const int c = (idx % 56) * 4;
    const long long e = tileBase + r;
    float4 v = make_float4(0.f, 0.f, 0.f, 0.f);
    if (e < (long long)E) {
      if (c < 64)       v = *(const float4*)(nf + (size_t)eidx[e] * 64 + c);
      else if (c < 128) v = *(const float4*)(nf + (size_t)eidx[(size_t)E + e] * 64 + (c - 64));
      else if (c < 192) v = *(const float4*)(nf + (size_t)eidx[2 * (size_t)E + e] * 64 + (c - 128));
      else              v = *(const float4*)(geo + (size_t)e * 32 + (c - 192));
    }
    uint2 p;
    p.x = (unsigned)f2bfu(v.x) | ((unsigned)f2bfu(v.y) << 16);
    p.y = (unsigned)f2bfu(v.z) | ((unsigned)f2bfu(v.w) << 16);
    *(uint2*)(sA + (size_t)r * TPAD + c) = p;
  }
  if (tid < 64) {
    const long long e = tileBase + tid;
    sMask[tid] = (e < (long long)E) ? (sf[e] ? 1 : 2) : 0;
  }
  __syncthreads();

  const int w = tid >> 5, lane = tid & 31;
  const int h = lane >> 4, n = lane & 15;

  v8f acc[2][4] = {};
#pragma unroll
  for (int kc = 0; kc < 7; ++kc) {
    // A fragment (16x32 bf16): lane<16 -> K in {h*8..h*8+7, 16+h*8..}
    const unsigned short* ap = sA + (size_t)(w * 16 + n) * TPAD + kc * 32 + h * 8;
    const v16bf afrag = make_frag(*(const uint4*)ap, *(const uint4*)(ap + 16));
#pragma unroll
    for (int br = 0; br < 2; ++br) {
#pragma unroll
      for (int nt = 0; nt < 4; ++nt) {
        // B fragment (32x16): lane holds K {h*8..h*8+7, 16+h*8..} for column nt*16+n
        const unsigned short* bp =
            W0b + (size_t)(br * 64 + nt * 16 + n) * FIN + kc * 32 + h * 8;
        const v16bf bfrag = make_frag(*(const uint4*)bp, *(const uint4*)(bp + 16));
        acc[br][nt] = __builtin_amdgcn_wmma_f32_16x16x32_bf16(
            false, afrag, false, bfrag, (short)0, acc[br][nt], false, false);
      }
    }
  }

  // epilogue: +bias, store pre-BN bf16, masked sum/sumsq
#pragma unroll
  for (int br = 0; br < 2; ++br) {
    unsigned short* X = br ? X1 : X0;
#pragma unroll
    for (int nt = 0; nt < 4; ++nt) {
      const int f = nt * 16 + n;
      const float bias = b0[br * 64 + f];
      float s = 0.f, s2 = 0.f;
#pragma unroll
      for (int v = 0; v < 8; ++v) {
        const int row = w * 16 + v + h * 8;       // C layout: M = v + h*8
        const float val = acc[br][nt][v] + bias;
        const int mk = sMask[row];
        if (mk) X[(size_t)(tileBase + row) * 64 + f] = f2bfu(val);
        if (mk == (br ? 2 : 1)) { s += val; s2 += val * val; }
      }
      s += __shfl_xor(s, 16);
      s2 += __shfl_xor(s2, 16);
      if (lane < 16) {
        atomicAdd(&sStats[br * 128 + f], s);
        atomicAdd(&sStats[br * 128 + 64 + f], s2);
      }
    }
  }
  __syncthreads();
  atomicAdd(&stats0[tid], sStats[tid]);
}

// ------------------- depth layers: BN+ReLU fused into staging --------------
__global__ __launch_bounds__(256) void spnn_gemmd(
    const unsigned short* __restrict__ Xin0, const unsigned short* __restrict__ Xin1,
    unsigned short* __restrict__ Xout0, unsigned short* __restrict__ Xout1,
    const unsigned short* __restrict__ Wl, int WbrStride,
    const float* __restrict__ bl, int bBrStride,
    const float* __restrict__ affL,   // input BN affine [2][A|B][64]
    float* __restrict__ statsOut,     // output stats   [2][sum|sq][64]
    const int* __restrict__ sf, int E) {
  __shared__ alignas(16) unsigned short sX[64 * XPAD];
  __shared__ unsigned char sMask[64];   // 0=invalid 1=this-branch 2=other
  __shared__ float sStats[128];

  const int tid = threadIdx.x;
  const int br = blockIdx.y;
  const long long tileBase = (long long)blockIdx.x * 64;
  const unsigned short* Xin = br ? Xin1 : Xin0;
  unsigned short* Xout = br ? Xout1 : Xout0;
  const float* Aaf = affL + br * 128;
  const unsigned short* Wb = Wl + (size_t)br * WbrStride;

  if (tid < 128) sStats[tid] = 0.f;

  // stage 64 rows x 64 bf16, apply input BN-affine + ReLU on the fly
#pragma unroll
  for (int it = 0; it < 2; ++it) {
    const int chunk = tid + it * 256;            // 512 chunks of 8 feats
    const int r = chunk >> 3, cq = chunk & 7;
    const long long e = tileBase + r;
    uint4 d = make_uint4(0, 0, 0, 0);
    if (e < (long long)E) d = *(const uint4*)(Xin + (size_t)e * 64 + cq * 8);
    unsigned pk[4];
    const unsigned* ds = &d.x;
#pragma unroll
    for (int q = 0; q < 4; ++q) {
      const int f0 = cq * 8 + q * 2;
      const float x0 = bf2f((unsigned short)(ds[q] & 0xffffu));
      const float x1 = bf2f((unsigned short)(ds[q] >> 16));
      const float y0 = fmaxf(x0 * Aaf[f0] + Aaf[64 + f0], 0.f);
      const float y1 = fmaxf(x1 * Aaf[f0 + 1] + Aaf[64 + f0 + 1], 0.f);
      pk[q] = (unsigned)f2bfu(y0) | ((unsigned)f2bfu(y1) << 16);
    }
    uint4 o; o.x = pk[0]; o.y = pk[1]; o.z = pk[2]; o.w = pk[3];
    *(uint4*)(sX + (size_t)r * XPAD + cq * 8) = o;
    if (cq == 0) {
      int m = 0;
      if (e < (long long)E) m = ((sf[e] != 0) == (br == 0)) ? 1 : 2;
      sMask[r] = (unsigned char)m;
    }
  }
  __syncthreads();

  const int w = tid >> 5, lane = tid & 31;
  const int h = lane >> 4, n = lane & 15;

  v8f acc[4] = {};
#pragma unroll
  for (int kc = 0; kc < 2; ++kc) {
    const unsigned short* ap = sX + (size_t)(w * 16 + n) * XPAD + kc * 32 + h * 8;
    const v16bf afrag = make_frag(*(const uint4*)ap, *(const uint4*)(ap + 16));
#pragma unroll
    for (int nt = 0; nt < 4; ++nt) {
      const unsigned short* bp = Wb + (size_t)(nt * 16 + n) * 64 + kc * 32 + h * 8;
      const v16bf bfrag = make_frag(*(const uint4*)bp, *(const uint4*)(bp + 16));
      acc[nt] = __builtin_amdgcn_wmma_f32_16x16x32_bf16(
          false, afrag, false, bfrag, (short)0, acc[nt], false, false);
    }
  }

#pragma unroll
  for (int nt = 0; nt < 4; ++nt) {
    const int f = nt * 16 + n;
    const float bias = bl[br * bBrStride + f];
    float s = 0.f, s2 = 0.f;
#pragma unroll
    for (int v = 0; v < 8; ++v) {
      const int row = w * 16 + v + h * 8;
      const float val = acc[nt][v] + bias;
      const int mk = sMask[row];
      if (mk) Xout[(size_t)(tileBase + row) * 64 + f] = f2bfu(val);
      if (mk == 1) { s += val; s2 += val * val; }
    }
    s += __shfl_xor(s, 16);
    s2 += __shfl_xor(s2, 16);
    if (lane < 16) { atomicAdd(&sStats[f], s); atomicAdd(&sStats[64 + f], s2); }
  }
  __syncthreads();
  if (tid < 128) atomicAdd(&statsOut[br * 128 + tid], sStats[tid]);
}

// --------------------- BN stats -> affine (A=g*rsqrt, B=be-mean*A) ---------
__global__ void spnn_finalize(const float* __restrict__ statsL,
                              const unsigned* __restrict__ cnt,
                              const float* __restrict__ gammaL,
                              const float* __restrict__ betaL,
                              int gbBrStride, float* __restrict__ affL, int E) {
  const int t = threadIdx.x;  // 128 threads
  const int br = t >> 6, f = t & 63;
  const unsigned c0 = cnt[0];
  float c = br ? (float)(E - (int)c0) : (float)c0;
  c = fmaxf(c, 1.f);
  const float mean = statsL[br * 128 + f] / c;
  const float var = statsL[br * 128 + 64 + f] / c - mean * mean;
  const float A = gammaL[br * gbBrStride + f] * rsqrtf(var + 1e-5f);
  affL[br * 128 + f] = A;
  affL[br * 128 + 64 + f] = betaL[br * gbBrStride + f] - mean * A;
}

// -------- final: BN+ReLU+leaky+att, masked branch select, scatter-add ------
__global__ __launch_bounds__(256) void spnn_scatter(
    const unsigned short* __restrict__ Xl0, const unsigned short* __restrict__ Xl1,
    const float* __restrict__ aff3, const float* __restrict__ att,
    const int* __restrict__ sf, const int* __restrict__ srcI,
    float* __restrict__ out, int E) {
  const long long idx = (long long)blockIdx.x * blockDim.x + threadIdx.x;
  if (idx >= (long long)E * 8) return;
  const long long e = idx >> 3;
  const int cq = (int)(idx & 7);
  const int br = (sf[e] != 0) ? 0 : 1;   // row belongs to exactly one branch
  const unsigned short* X = br ? Xl1 : Xl0;
  const uint4 d = *(const uint4*)(X + (size_t)e * 64 + cq * 8);
  const float a = att[br];
  const int node = srcI[e];
  const float* Af = aff3 + br * 128;
  const unsigned* ds = &d.x;
#pragma unroll
  for (int q = 0; q < 4; ++q) {
#pragma unroll
    for (int hl = 0; hl < 2; ++hl) {
      const int f = cq * 8 + q * 2 + hl;
      const float x = bf2f((unsigned short)(hl ? (ds[q] >> 16) : (ds[q] & 0xffffu)));
      float y = fmaxf(x * Af[f] + Af[64 + f], 0.f);
      y = (y > 0.f) ? y : 0.01f * y;     // leaky (identity post-ReLU, kept for fidelity)
      atomicAdd(out + (size_t)node * 64 + f, y * a);
    }
  }
}

// ---------------------------------------------------------------------------
extern "C" void kernel_launch(void* const* d_in, const int* in_sizes, int n_in,
                              void* d_out, int out_size, void* d_ws, size_t ws_size,
                              hipStream_t stream) {
  (void)n_in; (void)ws_size;
  const float* nf  = (const float*)d_in[0];
  const float* geo = (const float*)d_in[1];
  const int*   eidx = (const int*)d_in[2];   // [3][E]; row 0 = dest node i
  const float* att = (const float*)d_in[6];
  const int*   sf  = (const int*)d_in[7];
  const float* W0  = (const float*)d_in[8];
  const float* b0  = (const float*)d_in[9];
  const float* g0  = (const float*)d_in[10];
  const float* be0 = (const float*)d_in[11];
  const float* Wd  = (const float*)d_in[12];
  const float* bd  = (const float*)d_in[13];
  const float* gd  = (const float*)d_in[14];
  const float* bed = (const float*)d_in[15];
  float* out = (float*)d_out;
  const int E = in_sizes[7];

  // workspace carve-up (needs ~4*E*128B + ~120KB; ~1.02 GB for E=2M)
  char* p = (char*)d_ws;
  const size_t EH2 = (size_t)E * 64 * sizeof(unsigned short);
  auto alignup = [](size_t x) { return (x + 255) & ~(size_t)255; };
  size_t off = 0;
  unsigned short* A0 = (unsigned short*)(p + off); off += EH2;
  unsigned short* A1 = (unsigned short*)(p + off); off += EH2;
  unsigned short* B0 = (unsigned short*)(p + off); off += EH2;
  unsigned short* B1 = (unsigned short*)(p + off); off += EH2;
  off = alignup(off);
  unsigned short* W0b = (unsigned short*)(p + off); off += alignup(2 * 64 * FIN * 2);
  unsigned short* Wdb = (unsigned short*)(p + off); off += alignup(2 * 3 * 64 * 64 * 2);
  float*    stats = (float*)(p + off);    off += 4 * 256 * 4;  // 4 layers x [2][2][64]
  unsigned* cnt   = (unsigned*)(p + off); off += 256;
  float*    aff   = (float*)(p + off);    off += 4 * 256 * 4;

  hipMemsetAsync(out, 0, (size_t)out_size * sizeof(float), stream);
  hipMemsetAsync(stats, 0, 4 * 256 * 4 + 256, stream);  // stats + cnt (contiguous)

  spnn_convert_w<<<(2 * 64 * FIN + 2 * 3 * 64 * 64 + 255) / 256, 256, 0, stream>>>(W0, Wd, W0b, Wdb);
  spnn_count<<<512, 256, 0, stream>>>(sf, E, cnt);

  const int tiles = (E + 63) / 64;
  spnn_gemm0<<<tiles, 256, 0, stream>>>(nf, geo, eidx, sf, W0b, b0, A0, A1, stats, E);
  spnn_finalize<<<1, 128, 0, stream>>>(stats, cnt, g0, be0, 64, aff, E);

  unsigned short* pi0 = A0; unsigned short* pi1 = A1;
  unsigned short* po0 = B0; unsigned short* po1 = B1;
  for (int d = 0; d < 3; ++d) {
    spnn_gemmd<<<dim3(tiles, 2), 256, 0, stream>>>(
        pi0, pi1, po0, po1,
        Wdb + (size_t)d * 64 * 64, 3 * 64 * 64,
        bd + d * 64, 192,
        aff + d * 256, stats + (d + 1) * 256, sf, E);
    spnn_finalize<<<1, 128, 0, stream>>>(stats + (d + 1) * 256, cnt,
                                         gd + d * 64, bed + d * 64, 192,
                                         aff + (d + 1) * 256, E);
    unsigned short* t;
    t = pi0; pi0 = po0; po0 = t;
    t = pi1; pi1 = po1; po1 = t;
  }
  // last layer pre-BN activations are now in pi0/pi1
  const int nchunks = E * 8;
  spnn_scatter<<<(nchunks + 255) / 256, 256, 0, stream>>>(
      pi0, pi1, aff + 3 * 256, att, sf, eidx, out, E);
}
